// MASLoRALinear_17506286699159
// MI455X (gfx1250) — matrix operand order repaired and checked by reading, not compiled, not run on hardware
//
// MAS-LoRA Linear for MI455X (gfx1250), wave32 + WMMA bf16 + async LDS copies.
//
// Math: out[b] = x[b] @ (W_base + SCALING * sum_e w[b,e] B_e A_e)^T + b_base
// Folded into per-batch W_eff (rank-128 update), then one batched GEMM.
// Precision: bf16 operands, f32 accumulation via v_wmma_f32_16x16x32_bf16.
//
// ws layout (halfwords):
//   AT   : 1024*128           AstackT bf16
//   Bw   : 16*1024*128        gate-prescaled B factors bf16
//   Weff : 16*1024*1024       per-batch effective weights bf16
#include <hip/hip_runtime.h>

typedef __attribute__((ext_vector_type(16))) __bf16 v16bf;
typedef __attribute__((ext_vector_type(8)))  float  v8f;
typedef __attribute__((ext_vector_type(4)))  int    vi4;

typedef __attribute__((address_space(1))) vi4* gptr_b128;
typedef __attribute__((address_space(3))) vi4* lptr_b128;

#define SCALING 2.0f
#define Bn 16
#define Tn 1500
#define Cn 1024
#define On 1024
#define En 8
#define Rn 16
#define ERn 128

#define TM 128
#define TN 128
#define TK 64
#define PITCH 72   // halfwords per LDS row (64 data + 8 pad -> conflict-free b128 frags)

#if defined(__has_builtin)
#  if __has_builtin(__builtin_amdgcn_global_load_async_to_lds_b128)
#    define HAVE_ASYNC_LDS 1
#  else
#    define HAVE_ASYNC_LDS 0
#  endif
#else
#  define HAVE_ASYNC_LDS 0
#endif

__device__ __forceinline__ void wait_async0() {
#if HAVE_ASYNC_LDS
#  if __has_builtin(__builtin_amdgcn_s_wait_asynccnt)
  __builtin_amdgcn_s_wait_asynccnt(0);
#  else
  asm volatile("s_wait_asynccnt 0x0" ::: "memory");
#  endif
#endif
}

__device__ __forceinline__ unsigned short f2bf(float f) {
  unsigned u = __builtin_bit_cast(unsigned, f);
  u = u + 0x7FFFu + ((u >> 16) & 1u);   // round-to-nearest-even
  return (unsigned short)(u >> 16);
}

__device__ __forceinline__ v16bf make_frag(uint4 a, uint4 b) {
  struct P { uint4 x, y; } p{a, b};
  return __builtin_bit_cast(v16bf, p);
}

// Copy one 128x64-halfword tile global -> LDS ([row][k], pitch 72).
// Each wave moves 16 rows via 4 lane-wide b128 async ops (512B each).
__device__ __forceinline__ void tile_copy_async(unsigned short* ldst,
                                                const unsigned short* gsrc,
                                                int ld, int wave, int lane) {
  const int lrow = lane >> 3;   // 0..3
  const int seg  = lane & 7;    // 0..7, 8 halfwords each
#pragma unroll
  for (int j = 0; j < 4; ++j) {
    const int row = wave * 16 + j * 4 + lrow;
    const unsigned short* g = gsrc + (size_t)row * ld + seg * 8;
    unsigned short* d = ldst + row * PITCH + seg * 8;
#if HAVE_ASYNC_LDS
    __builtin_amdgcn_global_load_async_to_lds_b128(
        (gptr_b128)(void*)g, (lptr_b128)(void*)d, 0, 0);
#else
    uint4 v = *(const uint4*)g;
    *(uint4*)d = v;
#endif
  }
}

__device__ __forceinline__ void mma_stage(const unsigned short* lAp,
                                          const unsigned short* lBp,
                                          int wm, int wn, int l16, int halfsel,
                                          v8f acc[2][4]) {
#pragma unroll
  for (int ks = 0; ks < 2; ++ks) {           // two K=32 steps per TK=64 stage
    const int ksub = ks * 32;
    v16bf af[2], bfr[4];
#pragma unroll
    for (int mi = 0; mi < 2; ++mi) {
      // A frag: lane l -> row M=(l&15); half 0: K {0..7,16..23}, half 1: K {8..15,24..31}
      const unsigned short* p = lAp + (wm * 32 + mi * 16 + l16) * PITCH + ksub + halfsel * 8;
      af[mi] = make_frag(*(const uint4*)p, *(const uint4*)(p + 16));
    }
#pragma unroll
    for (int ni = 0; ni < 4; ++ni) {
      // B frag: lane l -> col N=(l&15); half 0: K {0..7,16..23}, half 1: K {8..15,24..31}
      const unsigned short* p = lBp + (wn * 64 + ni * 16 + l16) * PITCH + ksub + halfsel * 8;
      bfr[ni] = make_frag(*(const uint4*)p, *(const uint4*)(p + 16));
    }
#pragma unroll
    for (int mi = 0; mi < 2; ++mi)
#pragma unroll
      for (int ni = 0; ni < 4; ++ni)
        acc[mi][ni] = __builtin_amdgcn_wmma_f32_16x16x32_bf16(
            false, af[mi], false, bfr[ni], (short)0, acc[mi][ni], false, false);
  }
}

// 128x128-tile GEMM. A: f32 (staged + converted) or bf16 (async copy).
// B: always bf16 [N][K] row-major, async copy.
// EPI_MAIN: out f32 = D + b_base[n], rows guarded by Mvalid.
// !EPI_MAIN: out bf16 = D + addend[m*Cn+n]  (fold: += W_base).
template <bool A_F32, bool EPI_MAIN>
__global__ __launch_bounds__(256, 1) void wmma_gemm_kernel(
    const float* __restrict__ Af, const unsigned short* __restrict__ Ah, int lda,
    const unsigned short* __restrict__ Bh, int ldb,
    const float* __restrict__ addend,
    float* __restrict__ outF, unsigned short* __restrict__ outH,
    int K, int Mvalid,
    size_t strideA, size_t strideB, size_t strideOut) {
  const int tid = threadIdx.x;
  const int lane = tid & 31;
  const int wave = tid >> 5;
  const int wm = wave >> 1;      // 0..3 -> 32-row patch
  const int wn = wave & 1;       // 0..1 -> 64-col patch
  const int halfsel = lane >> 4;
  const int l16 = lane & 15;

  const int ntile = blockIdx.x, mtile = blockIdx.y, b = blockIdx.z;

  __shared__ unsigned short lA[2][TM * PITCH];
  __shared__ unsigned short lB[2][TN * PITCH];

  const unsigned short* Bbase = Bh + (size_t)b * strideB + (size_t)ntile * TN * ldb;

  // --- A f32 staging (main kernel only): 1 row x 32 floats per thread ---
  const int rowS = tid >> 1;
  const int colS = (tid & 1) * 32;
  const bool arow_ok = (mtile * TM + rowS) < Mvalid;
  float ra[32];

  auto loadA_f32 = [&](int kb) {
    const float* p = Af + (size_t)b * strideA +
                     (size_t)(mtile * TM + rowS) * lda + kb * TK + colS;
    if (arow_ok) {
#pragma unroll
      for (int i = 0; i < 8; ++i) {
        float4 v = *(const float4*)(p + i * 4);
        ra[i * 4 + 0] = v.x; ra[i * 4 + 1] = v.y;
        ra[i * 4 + 2] = v.z; ra[i * 4 + 3] = v.w;
      }
    } else {
#pragma unroll
      for (int i = 0; i < 32; ++i) ra[i] = 0.f;
    }
  };
  auto storeA_f32 = [&](int buf) {
    unsigned short* d = &lA[buf][rowS * PITCH + colS];
#pragma unroll
    for (int i = 0; i < 4; ++i) {
      uint4 q;
      q.x = (unsigned)f2bf(ra[i * 8 + 0]) | ((unsigned)f2bf(ra[i * 8 + 1]) << 16);
      q.y = (unsigned)f2bf(ra[i * 8 + 2]) | ((unsigned)f2bf(ra[i * 8 + 3]) << 16);
      q.z = (unsigned)f2bf(ra[i * 8 + 4]) | ((unsigned)f2bf(ra[i * 8 + 5]) << 16);
      q.w = (unsigned)f2bf(ra[i * 8 + 6]) | ((unsigned)f2bf(ra[i * 8 + 7]) << 16);
      ((uint4*)d)[i] = q;
    }
  };
  auto fillA = [&](int kb, int buf) {
    if constexpr (!A_F32) {
      const unsigned short* Abase = Ah + (size_t)b * strideA + (size_t)mtile * TM * lda;
      tile_copy_async(lA[buf], Abase + kb * TK, lda, wave, lane);
    }
  };
  auto fillB = [&](int kb, int buf) {
    tile_copy_async(lB[buf], Bbase + kb * TK, ldb, wave, lane);
  };

  v8f acc[2][4] = {};
  const int nstages = K / TK;

  // Prologue: fill buffer 0.
  fillB(0, 0);
  fillA(0, 0);
  if constexpr (A_F32) { loadA_f32(0); storeA_f32(0); }
  wait_async0();
  __syncthreads();

  int buf = 0;
  for (int kb = 0; kb < nstages; ++kb) {
    const bool more = (kb + 1 < nstages);
    if (more) {
      fillB(kb + 1, buf ^ 1);               // async -> LDS, no registers
      fillA(kb + 1, buf ^ 1);
      if constexpr (A_F32) loadA_f32(kb + 1);  // overlap global latency with MMA
    }
    mma_stage(lA[buf], lB[buf], wm, wn, l16, halfsel, acc);
    if (more && A_F32) storeA_f32(buf ^ 1);
    wait_async0();
    __syncthreads();
    buf ^= 1;
  }

  // Epilogue. D layout: VGPR i, lanes 0-15 -> M=i, lanes 16-31 -> M=i+8; N=(lane&15).
#pragma unroll
  for (int ni = 0; ni < 4; ++ni) {
    const int gn = ntile * TN + wn * 64 + ni * 16 + l16;
    float addv = 0.f;
    if constexpr (EPI_MAIN) addv = addend[gn];
#pragma unroll
    for (int mi = 0; mi < 2; ++mi) {
      const int mb = mtile * TM + wm * 32 + mi * 16 + halfsel * 8;
#pragma unroll
      for (int i = 0; i < 8; ++i) {
        const int gm = mb + i;
        const float v = acc[mi][ni][i];
        if constexpr (EPI_MAIN) {
          if (gm < Mvalid)
            outF[(size_t)b * strideOut + (size_t)gm * On + gn] = v + addv;
        } else {
          const float wb = addend[(size_t)gm * Cn + gn];
          outH[(size_t)b * strideOut + (size_t)gm * Cn + gn] = f2bf(v + wb);
        }
      }
    }
  }
}

// AstackT[c*128 + k] = bf16(As_flat[k*C + c])  (As (E,R,C) flattened over (e,r))
__global__ __launch_bounds__(256) void prep_astackT(const float* __restrict__ As,
                                                    unsigned short* __restrict__ AT) {
  int idx = blockIdx.x * 256 + threadIdx.x;
  if (idx >= Cn * ERn) return;
  int c = idx >> 7, k = idx & 127;
  AT[idx] = f2bf(As[(size_t)k * Cn + c]);
}

// Bw[b][o][k=16e+r] = bf16(SCALING * w[b,e] * Bs[e,o,r])
__global__ __launch_bounds__(256) void prep_bw(const float* __restrict__ Bs,
                                               const float* __restrict__ w,
                                               unsigned short* __restrict__ Bw) {
  int idx = blockIdx.x * 256 + threadIdx.x;
  if (idx >= Bn * On * ERn) return;
  int k = idx & 127, o = (idx >> 7) & 1023, b = idx >> 17;
  int e = k >> 4, r = k & 15;
  float v = SCALING * w[b * En + e] * Bs[((size_t)e * On + o) * Rn + r];
  Bw[idx] = f2bf(v);
}

extern "C" void kernel_launch(void* const* d_in, const int* in_sizes, int n_in,
                              void* d_out, int out_size, void* d_ws, size_t ws_size,
                              hipStream_t stream) {
  const float* x      = (const float*)d_in[0];
  const float* w      = (const float*)d_in[1];
  const float* W_base = (const float*)d_in[2];
  const float* b_base = (const float*)d_in[3];
  const float* As     = (const float*)d_in[4];
  const float* Bs     = (const float*)d_in[5];
  float* out = (float*)d_out;

  unsigned short* AT   = (unsigned short*)d_ws;                  // 1024*128
  unsigned short* Bw   = AT + (size_t)Cn * ERn;                  // 16*1024*128
  unsigned short* Weff = Bw + (size_t)Bn * On * ERn;             // 16*1024*1024

  prep_astackT<<<(Cn * ERn + 255) / 256, 256, 0, stream>>>(As, AT);
  prep_bw<<<(Bn * On * ERn + 255) / 256, 256, 0, stream>>>(Bs, w, Bw);

  // Fold: W_eff[b][o][c] = W_base[o][c] + Bw[b] (O x 128) @ AstackT^T (128 x C)
  dim3 gf(Cn / TN, On / TM, Bn);
  wmma_gemm_kernel<false, false><<<gf, 256, 0, stream>>>(
      nullptr, Bw, ERn, AT, ERn, W_base, nullptr, Weff,
      /*K=*/ERn, /*Mvalid=*/On,
      /*strideA=*/(size_t)On * ERn, /*strideB=*/0, /*strideOut=*/(size_t)On * Cn);

  // Main: out[b][t][o] = x[b] (T x C) @ W_eff[b]^T + b_base
  dim3 gm(On / TN, (Tn + TM - 1) / TM, Bn);
  wmma_gemm_kernel<true, true><<<gm, 256, 0, stream>>>(
      x, nullptr, Cn, Weff, Cn, b_base, out, nullptr,
      /*K=*/Cn, /*Mvalid=*/Tn,
      /*strideA=*/(size_t)Tn * Cn, /*strideB=*/(size_t)On * Cn,
      /*strideOut=*/(size_t)Tn * On);
}
